// SlotAttention_14998025797923
// MI455X (gfx1250) — matrix operand order repaired and checked by reading, not compiled, not run on hardware
//
#include <hip/hip_runtime.h>
#include <cstdint>

typedef __attribute__((ext_vector_type(16))) __bf16 v16bf;
typedef __attribute__((ext_vector_type(8)))  float  v8f;
typedef __attribute__((ext_vector_type(4)))  unsigned int u32x4;
typedef __attribute__((ext_vector_type(4)))  float  f32x4;
typedef __attribute__((ext_vector_type(4)))  int    i32x4;

#if defined(__AMDGCN__) && __has_builtin(__builtin_amdgcn_global_load_async_to_lds_b128) && __has_builtin(__builtin_amdgcn_s_wait_asynccnt)
#define HAVE_ASYNC_LDS 1
typedef i32x4 __attribute__((address_space(1)))* gptr_b128;
typedef i32x4 __attribute__((address_space(3)))* lptr_b128;
#endif

union Frag {            // one WMMA A/B operand: 16 bf16 = 32 bytes
  v16bf v;
  u32x4 q[2];
};

__device__ __forceinline__ uint16_t f2bf(float f) {
  union { float f; uint32_t u; } c{f};
  uint32_t u = c.u;
  u = u + 0x7FFFu + ((u >> 16) & 1u);   // round-to-nearest-even
  return (uint16_t)(u >> 16);
}
__device__ __forceinline__ float bf2f(uint16_t h) {
  union { uint32_t u; float f; } c{ (uint32_t)h << 16 };
  return c.f;
}

// ---------------------------------------------------------------------------
// slots = mu + exp(log_sigma) * noise
// ---------------------------------------------------------------------------
__global__ void init_slots_kernel(const float* __restrict__ noise,
                                  const float* __restrict__ mu,
                                  const float* __restrict__ logsig,
                                  float* __restrict__ slots, int n) {
  int i = blockIdx.x * blockDim.x + threadIdx.x;
  if (i < n) {
    int d = i & 63;
    slots[i] = mu[d] + __expf(logsig[d]) * noise[i];
  }
}

// ---------------------------------------------------------------------------
// Phase 1: x = LN(inputs); k = x@Wk; v = x@Wv   (bf16 outputs, WMMA bf16)
// 2048 blocks x 256 threads; block handles 128 rows (8 waves x 16 rows)
// ---------------------------------------------------------------------------
__global__ __launch_bounds__(256) void phase1_ln_kv(
    const float* __restrict__ inputs, const float* __restrict__ lng,
    const float* __restrict__ lnb, const float* __restrict__ Wk,
    const float* __restrict__ Wv, uint16_t* __restrict__ kbf,
    uint16_t* __restrict__ vbf)
{
  __shared__ __align__(16) uint16_t WkT[64][72];  // [d][din] bf16, padded
  __shared__ __align__(16) uint16_t WvT[64][72];
  __shared__ __align__(16) uint16_t xl[128][72];  // LN'd rows, bf16
  __shared__ float gg[64], bb[64];

  const int t = threadIdx.x;
  if (t < 64) { gg[t] = lng[t]; bb[t] = lnb[t]; }
  // transpose weights into LDS as bf16 (coalesced reads, padded scattered writes)
  for (int i = 0; i < 16; ++i) {
    int idx = t + i * 256;                 // 0..4095
    int kk = idx >> 6, d = idx & 63;
    WkT[d][kk] = f2bf(Wk[idx]);
    WvT[d][kk] = f2bf(Wv[idx]);
  }
  __syncthreads();

  const int wave = t >> 5, lane = t & 31;
  const int m = lane & 15, half = lane >> 4;
  const int row_local = wave * 16 + m;
  const long row = (long)blockIdx.x * 128 + row_local;

  // LayerNorm: 2 lanes per row, 32 elems each
  const float* xr = inputs + row * 64 + half * 32;
  float xv[32], s1 = 0.f, s2 = 0.f;
  for (int i = 0; i < 32; ++i) { float v = xr[i]; xv[i] = v; s1 += v; s2 += v * v; }
  s1 += __shfl_xor(s1, 16);
  s2 += __shfl_xor(s2, 16);
  const float mean = s1 * (1.f / 64.f);
  const float inv  = rsqrtf(s2 * (1.f / 64.f) - mean * mean + 1e-3f);

  uint32_t pk[16];
  for (int i = 0; i < 16; ++i) {
    int c0 = half * 32 + 2 * i;
    float y0 = (xv[2 * i]     - mean) * inv * gg[c0]     + bb[c0];
    float y1 = (xv[2 * i + 1] - mean) * inv * gg[c0 + 1] + bb[c0 + 1];
    pk[i] = (uint32_t)f2bf(y0) | ((uint32_t)f2bf(y1) << 16);
  }
  {
    u32x4* dst = (u32x4*)&xl[row_local][half * 32];
    dst[0] = u32x4{pk[0], pk[1], pk[2], pk[3]};
    dst[1] = u32x4{pk[4], pk[5], pk[6], pk[7]};
    dst[2] = u32x4{pk[8], pk[9], pk[10], pk[11]};
    dst[3] = u32x4{pk[12], pk[13], pk[14], pk[15]};
  }
  __syncthreads();

  // WMMA: each wave -> 16 rows x 64 cols of k and v
  const long out_base = ((long)blockIdx.x * 128 + wave * 16) * 64;
  const int dl    = lane & 15;
  const int abase = (lane < 16) ? 0 : 8;    // A K-run base (ISA layout)
  const int bbase = (lane < 16) ? 0 : 8;    // B K-run base
  const int rbase = (lane < 16) ? 0 : 8;    // C row base

  // hoist A fragments (shared by all four output col-tiles and by k & v)
  Frag a0, a1;
  a0.q[0] = *(const u32x4*)&xl[wave * 16 + m][abase];
  a0.q[1] = *(const u32x4*)&xl[wave * 16 + m][abase + 16];
  a1.q[0] = *(const u32x4*)&xl[wave * 16 + m][32 + abase];
  a1.q[1] = *(const u32x4*)&xl[wave * 16 + m][32 + abase + 16];

  for (int nt = 0; nt < 4; ++nt) {
    const int d = nt * 16 + dl;
    Frag bk0, bk1, bv0, bv1;
    bk0.q[0] = *(const u32x4*)&WkT[d][bbase];
    bk0.q[1] = *(const u32x4*)&WkT[d][bbase + 16];
    bk1.q[0] = *(const u32x4*)&WkT[d][32 + bbase];
    bk1.q[1] = *(const u32x4*)&WkT[d][32 + bbase + 16];
    bv0.q[0] = *(const u32x4*)&WvT[d][bbase];
    bv0.q[1] = *(const u32x4*)&WvT[d][bbase + 16];
    bv1.q[0] = *(const u32x4*)&WvT[d][32 + bbase];
    bv1.q[1] = *(const u32x4*)&WvT[d][32 + bbase + 16];
    v8f ack = {}, acv = {};
    ack = __builtin_amdgcn_wmma_f32_16x16x32_bf16(false, a0.v, false, bk0.v,
                                                  (short)0, ack, false, false);
    ack = __builtin_amdgcn_wmma_f32_16x16x32_bf16(false, a1.v, false, bk1.v,
                                                  (short)0, ack, false, false);
    acv = __builtin_amdgcn_wmma_f32_16x16x32_bf16(false, a0.v, false, bv0.v,
                                                  (short)0, acv, false, false);
    acv = __builtin_amdgcn_wmma_f32_16x16x32_bf16(false, a1.v, false, bv1.v,
                                                  (short)0, acv, false, false);
    for (int r = 0; r < 8; ++r) {
      long o = out_base + (long)(r + rbase) * 64 + d;
      kbf[o] = f2bf(ack[r]);
      vbf[o] = f2bf(acv[r]);
    }
  }
}

// ---------------------------------------------------------------------------
// q = LN(slots) @ Wq * D^-0.5    (tiny; 64 blocks x 64 threads)
// ---------------------------------------------------------------------------
__global__ __launch_bounds__(64) void q_kernel(
    const float* __restrict__ slots, const float* __restrict__ g,
    const float* __restrict__ bt, const float* __restrict__ Wq,
    float* __restrict__ qout)
{
  __shared__ float sl[7][64], lnv[7][64], mean[7], inv[7];
  const int b = blockIdx.x, t = threadIdx.x;
  for (int s = 0; s < 7; ++s) sl[s][t] = slots[(b * 7 + s) * 64 + t];
  __syncthreads();
  if (t < 7) {
    float a1 = 0.f, a2 = 0.f;
    for (int k = 0; k < 64; ++k) { float v = sl[t][k]; a1 += v; a2 += v * v; }
    float mn = a1 / 64.f;
    mean[t] = mn;
    inv[t] = rsqrtf(a2 / 64.f - mn * mn + 1e-3f);
  }
  __syncthreads();
  for (int s = 0; s < 7; ++s) lnv[s][t] = (sl[s][t] - mean[s]) * inv[s] * g[t] + bt[t];
  __syncthreads();
  for (int s = 0; s < 7; ++s) {
    float acc = 0.f;
    for (int k = 0; k < 64; ++k) acc += lnv[s][k] * Wq[k * 64 + t];
    qout[(b * 7 + s) * 64 + t] = acc * 0.125f;   // 64^-0.5
  }
}

__global__ void zero_kernel(float* __restrict__ U, float* __restrict__ cs) {
  int i = blockIdx.x * blockDim.x + threadIdx.x;
  if (i < 64 * 7 * 64) U[i] = 0.f;
  if (i < 64 * 7) cs[i] = 0.f;
}

// ---------------------------------------------------------------------------
// Attention: logits = k@qT (WMMA), softmax over S, p (+eps) kept un-normalized;
// accumulate U_raw[b,s,d] = sum_n p*v and colsum[b,s] = sum_n p via atomics.
// grid = B*16 blocks (256 N-rows per block) x 256 threads
// ---------------------------------------------------------------------------
__global__ __launch_bounds__(256) void attn_kernel(
    const uint16_t* __restrict__ kbf, const uint16_t* __restrict__ vbf,
    const float* __restrict__ qf, float* __restrict__ U,
    float* __restrict__ colsum)
{
  __shared__ __align__(16) uint16_t ql[16][64];    // q bf16, rows 7..15 zero
  __shared__ __align__(16) float    pT[16][264];   // p transposed [s][n]
  __shared__ __align__(16) uint16_t vl[256][72];   // v tile bf16

  const int b = blockIdx.x >> 4, chunk = blockIdx.x & 15;
  const long row0 = (long)b * 4096 + chunk * 256;
  const int t = threadIdx.x;

#ifdef HAVE_ASYNC_LDS
  // Async DMA the v tile straight into LDS (ASYNCcnt path, no VGPR round trip);
  // overlaps with the q staging below. Row stride in LDS is 144B, so issue
  // eight b128 moves per row with explicit pointers.
  {
    gptr_b128 gp = (gptr_b128)(vbf + (row0 + t) * 64);
    lptr_b128 lp = (lptr_b128)&vl[t][0];
#pragma unroll
    for (int j = 0; j < 8; ++j)
      __builtin_amdgcn_global_load_async_to_lds_b128(gp + j, lp + j, 0, 0);
  }
#else
  {  // fallback: stage v tile through VGPRs; thread t copies row t (128B)
    const u32x4* src = (const u32x4*)(vbf + row0 * 64);
    u32x4* dst = (u32x4*)&vl[t][0];
    for (int j = 0; j < 8; ++j) dst[j] = src[t * 8 + j];
  }
#endif

  for (int i = t; i < 16 * 64; i += 256) {
    int s = i >> 6, c = i & 63;
    ql[s][c] = (s < 7) ? f2bf(qf[(b * 7 + s) * 64 + c]) : (uint16_t)0;
  }

#ifdef HAVE_ASYNC_LDS
  __builtin_amdgcn_s_wait_asynccnt(0);   // drain own DMAs before barrier
#endif
  __syncthreads();

  const int wave = t >> 5, lane = t & 31;
  const int sidx  = lane & 15;             // A row / B col / C col
  const int abase = (lane < 16) ? 0 : 8;
  const int bbase = (lane < 16) ? 0 : 8;
  const int rbase = (lane < 16) ? 0 : 8;

  for (int tile = 0; tile < 2; ++tile) {
    const int rloc = wave * 32 + tile * 16;
    const uint16_t* krow = kbf + (row0 + rloc + sidx) * 64;
    v8f acc = {};
    for (int kt = 0; kt < 2; ++kt) {
      Frag a, bq;
      a.q[0]  = *(const u32x4*)(krow + kt * 32 + abase);
      a.q[1]  = *(const u32x4*)(krow + kt * 32 + abase + 16);
      bq.q[0] = *(const u32x4*)&ql[sidx][kt * 32 + bbase];
      bq.q[1] = *(const u32x4*)&ql[sidx][kt * 32 + bbase + 16];
      acc = __builtin_amdgcn_wmma_f32_16x16x32_bf16(false, a.v, false, bq.v,
                                                    (short)0, acc, false, false);
    }
    // softmax over slots: each row lives in 16 lanes (col = sidx)
    const bool valid = sidx < 7;
    float pf[8];
    for (int r = 0; r < 8; ++r) {
      float vlog = valid ? acc[r] : -3.0e38f;
      float mx = vlog;
      mx = fmaxf(mx, __shfl_xor(mx, 1));
      mx = fmaxf(mx, __shfl_xor(mx, 2));
      mx = fmaxf(mx, __shfl_xor(mx, 4));
      mx = fmaxf(mx, __shfl_xor(mx, 8));
      float e = valid ? __expf(vlog - mx) : 0.f;
      float se = e;
      se += __shfl_xor(se, 1);
      se += __shfl_xor(se, 2);
      se += __shfl_xor(se, 4);
      se += __shfl_xor(se, 8);
      pf[r] = valid ? (e / se + 1e-8f) : 0.f;
    }
    f32x4* pd = (f32x4*)&pT[sidx][rloc + rbase];   // 8 consecutive n
    pd[0] = f32x4{pf[0], pf[1], pf[2], pf[3]};
    pd[1] = f32x4{pf[4], pf[5], pf[6], pf[7]};
  }
  __syncthreads();

  // U_raw / colsum accumulation (VALU out of LDS; only ~0.23 GFLOP/iter total)
  const int s = t >> 5;
  if (s < 7) {
    const int d0 = (t & 31) * 2;
    float a0 = 0.f, a1 = 0.f;
    for (int n = 0; n < 256; ++n) {
      float pv = pT[s][n];
      uint32_t vv = *(const uint32_t*)&vl[n][d0];   // two bf16 in one DS load
      a0 += pv * bf2f((uint16_t)vv);
      a1 += pv * bf2f((uint16_t)(vv >> 16));
    }
    atomicAdd(&U[(b * 7 + s) * 64 + d0],     a0);
    atomicAdd(&U[(b * 7 + s) * 64 + d0 + 1], a1);
    if ((t & 31) == 0) {
      float cs = 0.f;
      for (int n = 0; n < 256; ++n) cs += pT[s][n];
      atomicAdd(&colsum[b * 7 + s], cs);
    }
  }
}

// ---------------------------------------------------------------------------
// updates = U/colsum; GRU (reset_after); slots += MLP(LN(slots))
// 64 blocks x 192 threads
// ---------------------------------------------------------------------------
__global__ __launch_bounds__(192) void update_kernel(
    const float* __restrict__ U, const float* __restrict__ colsum,
    const float* __restrict__ slots_in, const float* __restrict__ gW,
    const float* __restrict__ gU, const float* __restrict__ gb,
    const float* __restrict__ mg, const float* __restrict__ mb,
    const float* __restrict__ W1, const float* __restrict__ b1,
    const float* __restrict__ W2, const float* __restrict__ b2,
    float* __restrict__ slots_out, float* __restrict__ out)
{
  __shared__ float upd[7][64], sp[7][64], mx[7][192], mh[7][192];
  __shared__ float sn[7][64], mean[7], inv[7], h1[7][128];
  const int b = blockIdx.x, t = threadIdx.x;

  for (int i = t; i < 448; i += 192) {
    int s = i >> 6, d = i & 63;
    upd[s][d] = U[(b * 7 + s) * 64 + d] / colsum[b * 7 + s];
    sp[s][d]  = slots_in[(b * 7 + s) * 64 + d];
  }
  __syncthreads();
  {
    const float bx = gb[t], bh = gb[192 + t];
    for (int s = 0; s < 7; ++s) {
      float ax = bx, ah = bh;
      for (int k = 0; k < 64; ++k) {
        ax += upd[s][k] * gW[k * 192 + t];
        ah += sp[s][k]  * gU[k * 192 + t];
      }
      mx[s][t] = ax;
      mh[s][t] = ah;
    }
  }
  __syncthreads();
  for (int i = t; i < 448; i += 192) {
    int s = i >> 6, d = i & 63;
    float z = 1.f / (1.f + __expf(-(mx[s][d]      + mh[s][d])));
    float r = 1.f / (1.f + __expf(-(mx[s][64 + d] + mh[s][64 + d])));
    float hc = tanhf(mx[s][128 + d] + r * mh[s][128 + d]);
    sn[s][d] = z * sp[s][d] + (1.f - z) * hc;
  }
  __syncthreads();
  if (t < 7) {
    float a1 = 0.f, a2 = 0.f;
    for (int k = 0; k < 64; ++k) { float v = sn[t][k]; a1 += v; a2 += v * v; }
    float mn = a1 / 64.f;
    mean[t] = mn;
    inv[t] = rsqrtf(a2 / 64.f - mn * mn + 1e-3f);
  }
  __syncthreads();
  if (t < 128) {
    for (int s = 0; s < 7; ++s) {
      float acc = b1[t];
      for (int k = 0; k < 64; ++k)
        acc += ((sn[s][k] - mean[s]) * inv[s] * mg[k] + mb[k]) * W1[k * 128 + t];
      h1[s][t] = fmaxf(acc, 0.f);
    }
  }
  __syncthreads();
  for (int i = t; i < 448; i += 192) {
    int s = i >> 6, d = i & 63;
    float acc = sn[s][d] + b2[d];
    for (int j = 0; j < 128; ++j) acc += h1[s][j] * W2[j * 64 + d];
    slots_out[(b * 7 + s) * 64 + d] = acc;
    out[(b * 7 + s) * 64 + d] = acc;
  }
}

// ---------------------------------------------------------------------------
extern "C" void kernel_launch(void* const* d_in, const int* in_sizes, int n_in,
                              void* d_out, int out_size, void* d_ws, size_t ws_size,
                              hipStream_t stream) {
  const float* inputs   = (const float*)d_in[0];
  const float* noise    = (const float*)d_in[1];
  const float* ln_in_g  = (const float*)d_in[2];
  const float* ln_in_b  = (const float*)d_in[3];
  const float* ln_s_g   = (const float*)d_in[4];
  const float* ln_s_b   = (const float*)d_in[5];
  const float* ln_m_g   = (const float*)d_in[6];
  const float* ln_m_b   = (const float*)d_in[7];
  const float* Wq       = (const float*)d_in[8];
  const float* Wk       = (const float*)d_in[9];
  const float* Wv       = (const float*)d_in[10];
  const float* mu       = (const float*)d_in[11];
  const float* logsig   = (const float*)d_in[12];
  const float* gru_W    = (const float*)d_in[13];
  const float* gru_U    = (const float*)d_in[14];
  const float* gru_b    = (const float*)d_in[15];
  const float* W1       = (const float*)d_in[16];
  const float* b1       = (const float*)d_in[17];
  const float* W2       = (const float*)d_in[18];
  const float* b2       = (const float*)d_in[19];
  float* out = (float*)d_out;

  // workspace carve: kbf 32MB | vbf 32MB | slots | q | U | colsum
  char* ws = (char*)d_ws;
  uint16_t* kbf = (uint16_t*)ws;
  uint16_t* vbf = (uint16_t*)(ws + (size_t)32 * 1024 * 1024);
  float* slots  = (float*)(ws + (size_t)64 * 1024 * 1024);
  float* qbuf   = slots + 28672;
  float* Ubuf   = qbuf + 28672;
  float* csum   = Ubuf + 28672;

  init_slots_kernel<<<112, 256, 0, stream>>>(noise, mu, logsig, slots, 28672);
  phase1_ln_kv<<<2048, 256, 0, stream>>>(inputs, ln_in_g, ln_in_b, Wk, Wv, kbf, vbf);

  for (int it = 0; it < 3; ++it) {
    q_kernel<<<64, 64, 0, stream>>>(slots, ln_s_g, ln_s_b, Wq, qbuf);
    zero_kernel<<<112, 256, 0, stream>>>(Ubuf, csum);
    attn_kernel<<<1024, 256, 0, stream>>>(kbf, vbf, qbuf, Ubuf, csum);
    update_kernel<<<64, 192, 0, stream>>>(Ubuf, csum, slots, gru_W, gru_U, gru_b,
                                          ln_m_g, ln_m_b, W1, b1, W2, b2,
                                          slots, out);
  }
}
